// channel_attention_62801011802641
// MI455X (gfx1250) — matrix operand order repaired and checked by reading, not compiled, not run on hardware
//
#include <hip/hip_runtime.h>

typedef __attribute__((ext_vector_type(16))) _Float16 v16h;
typedef __attribute__((ext_vector_type(8)))  _Float16 v8h;
typedef __attribute__((ext_vector_type(8)))  float    v8f;
typedef __attribute__((ext_vector_type(2)))  float    v2f;

#define B_      32
#define D_      8
#define C_      64
#define T_      3000
#define INTER_  30
#define EXT_    100
#define BD_     (B_ * D_)
#define TILE_T  120      // 4 pooling windows per tile; 25 tiles cover T
#define NPAD    128      // tile padded to WMMA N multiple of 16
#define ZLD     129      // Z row stride (floats): conflict-free column walks
#define FRG_LD  72       // fragment-matrix row stride in halves (144B): b128 gathers conflict-free
#define LN_EPS  1e-5f

// Map natural K (0..63) to fragment-ready position within a 64-wide row so a
// lane's 16 A-fragment halves are contiguous:
//   frag element j (0..15) of (kblk, hi) <-> K = kblk*32 + (j<8 ? hi*8+j : 16+hi*8+j-8)
__device__ __forceinline__ int wfrag_pos(int K) {
  const int kblk = K >> 5;
  const int r    = K & 31;
  const int hi   = (r & 15) >> 3;
  const int j    = (r & 7) + ((r & 16) ? 8 : 0);
  return kblk * 32 + hi * 16 + j;
}

__device__ __forceinline__ v16h load_frag(const _Float16* p) {
  union { v16h v; v8h h[2]; } u;
  u.h[0] = *(const v8h*)(p);
  u.h[1] = *(const v8h*)(p + 8);
  return u.v;
}

// ---------------------------------------------------------------------------
// 8 wave32 compute Z[64][NPAD] = W(64x64) @ X(64xNPAD) + bias via
// v_wmma_f32_16x16x32_f16.  Wswz: [64][FRG_LD] K-permuted rows (A-frag order).
// Xt: [NPAD][FRG_LD] column-major tile (B-frag order: same K permutation as Wswz).
// Wave w owns M-block (w&3), N-blocks (w>>2)*4 .. +3.
// ---------------------------------------------------------------------------
__device__ __forceinline__ void wave_gemm_64xN(const _Float16* __restrict__ Wswz,
                                               const _Float16* __restrict__ Xt,
                                               const float* __restrict__ bias,
                                               float* __restrict__ Zs)
{
  const int tid  = threadIdx.x;
  const int wave = tid >> 5;
  const int lane = tid & 31;
  const int lo   = lane & 15;
  const int hi   = lane >> 4;
  const int mi   = wave & 3;
  const int nhalf= wave >> 2;

  const _Float16* wrow = Wswz + (mi * 16 + lo) * FRG_LD + hi * 16;
  const v16h a0 = load_frag(wrow);        // K = 0..31 (this lane's share)
  const v16h a1 = load_frag(wrow + 32);   // K = 32..63

#pragma unroll
  for (int j = 0; j < 4; ++j) {
    const int nb = nhalf * 4 + j;
    const _Float16* xrow = Xt + (nb * 16 + lo) * FRG_LD + hi * 16;
    const v16h b0 = load_frag(xrow);      // K = hi*8 + 0..7, 16 + hi*8 + 0..7
    const v16h b1 = load_frag(xrow + 32); // K = 32 + {hi*8, 16 + hi*8} + 0..7
    v8f acc = {};
    acc = __builtin_amdgcn_wmma_f32_16x16x32_f16(false, a0, false, b0, (short)0, acc, false, false);
    acc = __builtin_amdgcn_wmma_f32_16x16x32_f16(false, a1, false, b1, (short)0, acc, false, false);
#pragma unroll
    for (int r = 0; r < 8; ++r) {
      const int row = mi * 16 + hi * 8 + r;   // C/D: VGPR r -> M = hi*8 + r, N = lo
      Zs[row * ZLD + nb * 16 + lo] = acc[r] + bias[row];
    }
  }
}

// LayerNorm over 64 channels of each column (thread = one column), 2 LDS passes.
__device__ __forceinline__ void col_layernorm(float* __restrict__ Zs,
                                              const float* __restrict__ g,
                                              const float* __restrict__ beta,
                                              int ncols)
{
  const int col = threadIdx.x;
  if (col < ncols) {
    float s = 0.f, ss = 0.f;
#pragma unroll
    for (int c = 0; c < 64; ++c) { const float z = Zs[c * ZLD + col]; s += z; ss += z * z; }
    const float mu  = s * (1.0f / 64.0f);
    float var = ss * (1.0f / 64.0f) - mu * mu;
    var = fmaxf(var, 0.f);
    const float rs = rsqrtf(var + LN_EPS);
#pragma unroll
    for (int c = 0; c < 64; ++c) {
      const float z = Zs[c * ZLD + col];
      Zs[c * ZLD + col] = (z - mu) * rs * g[c] + beta[c];
    }
  }
}

// ---------------------------------------------------------------------------
// Kernel 1: q/k linear + LN + avg-pool(30) -> qp,kp [BD][64][100]
// ---------------------------------------------------------------------------
__global__ __launch_bounds__(256)
void qk_pool_kernel(const float* __restrict__ x,
                    const float* __restrict__ qW, const float* __restrict__ qb,
                    const float* __restrict__ qg, const float* __restrict__ qbe,
                    const float* __restrict__ kW, const float* __restrict__ kb,
                    const float* __restrict__ kg, const float* __restrict__ kbe,
                    float* __restrict__ qp, float* __restrict__ kp)
{
  extern __shared__ char smem[];
  float*    Zs   = (float*)smem;                         // [64*129] f32  (33024 B)
  float*    vecs = (float*)(smem + 33024);               // 6*64 f32      (1536 B)
  float *bq = vecs, *gq = vecs + 64, *bbq = vecs + 128;
  float *bk = vecs + 192, *gk = vecs + 256, *bbk = vecs + 320;
  _Float16* Wq = (_Float16*)(smem + 34560);              // [64][72] f16  (9216 B)
  _Float16* Wk = (_Float16*)(smem + 43776);              // [64][72] f16  (9216 B)
  _Float16* Xt = (_Float16*)(smem + 52992);              // [128][72] f16 (18432 B)

  const int tid  = threadIdx.x;
  const int tile = blockIdx.x;              // 0..24
  const int bd   = blockIdx.y;              // 0..255
  const int t0   = tile * TILE_T;

  for (int i = tid; i < 4096; i += 256) {
    const int m = i >> 6, K = i & 63;
    const int p = m * FRG_LD + wfrag_pos(K);
    Wq[p] = (_Float16)qW[i];
    Wk[p] = (_Float16)kW[i];
  }
  if (tid < 64) {
    bq[tid] = qb[tid]; gq[tid] = qg[tid]; bbq[tid] = qbe[tid];
    bk[tid] = kb[tid]; gk[tid] = kg[tid]; bbk[tid] = kbe[tid];
  }
  for (int i = tid; i < 64 * NPAD; i += 256) {
    const int c = i >> 7, tc = i & 127;
    const float v = (tc < TILE_T) ? x[(size_t)(bd * 64 + c) * T_ + t0 + tc] : 0.f;
    Xt[tc * FRG_LD + wfrag_pos(c)] = (_Float16)v;      // transpose: row = column index n
  }
  __syncthreads();

  // ---- Q path ----
  wave_gemm_64xN(Wq, Xt, bq, Zs);
  __syncthreads();
  col_layernorm(Zs, gq, bbq, TILE_T);
  __syncthreads();
  {
    const int c = tid & 63, wl = tid >> 6;   // 64 channels x 4 windows
    float s = 0.f;
    for (int j = 0; j < INTER_; ++j) s += Zs[c * ZLD + wl * INTER_ + j];
    qp[(size_t)(bd * 64 + c) * EXT_ + tile * 4 + wl] = s * (1.0f / INTER_);
  }
  __syncthreads();

  // ---- K path (reuse Zs) ----
  wave_gemm_64xN(Wk, Xt, bk, Zs);
  __syncthreads();
  col_layernorm(Zs, gk, bbk, TILE_T);
  __syncthreads();
  {
    const int c = tid & 63, wl = tid >> 6;
    float s = 0.f;
    for (int j = 0; j < INTER_; ++j) s += Zs[c * ZLD + wl * INTER_ + j];
    kp[(size_t)(bd * 64 + c) * EXT_ + tile * 4 + wl] = s * (1.0f / INTER_);
  }
}

// ---------------------------------------------------------------------------
// Kernel 2: att = qp @ kp^T / 10 via v_wmma_f32_16x16x4_f32, softmax over m,
// S[c] = sum_m score[c][m].
// ---------------------------------------------------------------------------
__global__ __launch_bounds__(256)
void att_softmax_kernel(const float* __restrict__ qp, const float* __restrict__ kp,
                        float* __restrict__ S)
{
  __shared__ float att[64][68];
  const int tid  = threadIdx.x;
  const int bd   = blockIdx.x;
  const int wave = tid >> 5, lane = tid & 31, lo = lane & 15, hi = lane >> 4;
  const int mi    = wave >> 1;
  const int nbase = (wave & 1) * 2;

  const float* qrow = qp + (size_t)(bd * 64) * EXT_;
  const float* krow = kp + (size_t)(bd * 64) * EXT_;

#pragma unroll
  for (int j = 0; j < 2; ++j) {
    const int nb = nbase + j;
    v8f acc = {};
    for (int ks = 0; ks < 25; ++ks) {
      // 32-bit A 16x4: lanes 0-15 K=0,1 ; lanes 16-31 K=2,3 ; M = lo
      // 32-bit B 4x16: same K striping, N = lo ; B = kp^T so B[e][m] = kp[m][e]
      const v2f a = *(const v2f*)(qrow + (mi * 16 + lo) * EXT_ + ks * 4 + hi * 2);
      const v2f b = *(const v2f*)(krow + (nb * 16 + lo) * EXT_ + ks * 4 + hi * 2);
      acc = __builtin_amdgcn_wmma_f32_16x16x4_f32(false, a, false, b, (short)0, acc, false, false);
    }
#pragma unroll
    for (int r = 0; r < 8; ++r)
      att[mi * 16 + hi * 8 + r][nb * 16 + lo] = acc[r] * 0.1f;   // / sqrt(EXT)
  }
  __syncthreads();

  if (tid < 64) {
    float mx = -3.402823466e38f;
    for (int m = 0; m < 64; ++m) mx = fmaxf(mx, att[tid][m]);
    float se = 0.f;
    for (int m = 0; m < 64; ++m) se += expf(att[tid][m] - mx);
    float s = 0.f;                       // faithful: sum of individually-divided probs
    for (int m = 0; m < 64; ++m) s += expf(att[tid][m] - mx) / se;
    S[bd * 64 + tid] = s;
  }
}

// ---------------------------------------------------------------------------
// Kernel 3: out = LN( pW @ (x * S[c]) + pb )  written in [b,d,c,t] layout
// ---------------------------------------------------------------------------
__global__ __launch_bounds__(256)
void proj_out_kernel(const float* __restrict__ x, const float* __restrict__ S,
                     const float* __restrict__ pW, const float* __restrict__ pb,
                     const float* __restrict__ pg, const float* __restrict__ pbe,
                     float* __restrict__ out)
{
  extern __shared__ char smem[];
  float*    Zs   = (float*)smem;                         // [64*129] f32  (33024 B)
  float*    vecs = (float*)(smem + 33024);               // 4*64 f32      (1024 B)
  float *bp = vecs, *gp = vecs + 64, *bbp = vecs + 128, *sS = vecs + 192;
  _Float16* Wp = (_Float16*)(smem + 34048);              // [64][72] f16  (9216 B)
  _Float16* Xt = (_Float16*)(smem + 43264);              // [128][72] f16 (18432 B)

  const int tid  = threadIdx.x;
  const int tile = blockIdx.x;
  const int bd   = blockIdx.y;
  const int t0   = tile * TILE_T;

  for (int i = tid; i < 4096; i += 256) {
    const int m = i >> 6, K = i & 63;
    Wp[m * FRG_LD + wfrag_pos(K)] = (_Float16)pW[i];
  }
  if (tid < 64) { bp[tid] = pb[tid]; gp[tid] = pg[tid]; bbp[tid] = pbe[tid]; sS[tid] = S[bd * 64 + tid]; }
  __syncthreads();
  for (int i = tid; i < 64 * NPAD; i += 256) {
    const int c = i >> 7, tc = i & 127;
    const float v = (tc < TILE_T) ? x[(size_t)(bd * 64 + c) * T_ + t0 + tc] * sS[c] : 0.f;
    Xt[tc * FRG_LD + wfrag_pos(c)] = (_Float16)v;
  }
  __syncthreads();

  wave_gemm_64xN(Wp, Xt, bp, Zs);
  __syncthreads();
  col_layernorm(Zs, gp, bbp, TILE_T);
  __syncthreads();

  for (int i = tid; i < 64 * TILE_T; i += 256) {
    const int c = i / TILE_T, t = i - c * TILE_T;
    out[(size_t)(bd * 64 + c) * T_ + t0 + t] = Zs[c * ZLD + t];
  }
}

// ---------------------------------------------------------------------------
extern "C" void kernel_launch(void* const* d_in, const int* in_sizes, int n_in,
                              void* d_out, int out_size, void* d_ws, size_t ws_size,
                              hipStream_t stream)
{
  (void)in_sizes; (void)n_in; (void)out_size; (void)ws_size;
  const float* x   = (const float*)d_in[0];
  const float* qW  = (const float*)d_in[1];
  const float* qb  = (const float*)d_in[2];
  const float* qg  = (const float*)d_in[3];
  const float* qbe = (const float*)d_in[4];
  const float* kW  = (const float*)d_in[5];
  const float* kb  = (const float*)d_in[6];
  const float* kg  = (const float*)d_in[7];
  const float* kbe = (const float*)d_in[8];
  const float* pW  = (const float*)d_in[9];
  const float* pb  = (const float*)d_in[10];
  const float* pg  = (const float*)d_in[11];
  const float* pbe = (const float*)d_in[12];
  float* out = (float*)d_out;

  float* qp = (float*)d_ws;                        // [BD][64][100]
  float* kp = qp + (size_t)BD_ * 64 * EXT_;        // [BD][64][100]
  float* S  = kp + (size_t)BD_ * 64 * EXT_;        // [BD][64]

  const size_t smem1 = 52992 + 18432;   // Zs+vecs+Wq+Wk+Xt  = 71424 B
  const size_t smem3 = 43264 + 18432;   // Zs+vecs+Wp+Xt     = 61696 B

  dim3 grid1(T_ / TILE_T, BD_);   // 25 x 256
  qk_pool_kernel<<<grid1, 256, smem1, stream>>>(x, qW, qb, qg, qbe, kW, kb, kg, kbe, qp, kp);
  att_softmax_kernel<<<BD_, 256, 0, stream>>>(qp, kp, S);
  proj_out_kernel<<<grid1, 256, smem3, stream>>>(x, S, pW, pb, pg, pbe, out);
}
